// IncrGRU_89644557402351
// MI455X (gfx1250) — matrix operand order repaired and checked by reading, not compiled, not run on hardware
//
#include <hip/hip_runtime.h>

#define BATCH  131072
#define INF    40
#define PP     128
#define HH     256

// WG tile: 64 batch rows, 128 threads = 4 waves, 16 rows per wave.
#define BT     64
#define NTHR   128

typedef __attribute__((ext_vector_type(16))) __bf16 v16bf;
typedef __attribute__((ext_vector_type(8)))  float  v8f;
typedef __attribute__((ext_vector_type(4)))  __bf16 bf4;
typedef __attribute__((ext_vector_type(4)))  int    v4i;

// LDS row strides in bf16 elements (+8 pad: stride = 4 dwords mod 64 banks (XA_LD: 36), 16B aligned)
#define XA_LD 72     // staged x, K padded 40 -> 64
#define HA_LD 264    // staged h, K = 256
#define XP_LD 136    // staged x_proj, K = 128
#define WI_LD 136    // staged W_ih rows, K = 128
#define WH_LD 264    // staged W_hh rows, K = 256
#define WIB_SZ (3 * 16 * WI_LD)   // 6528 bf16 per buffer
#define WHB_SZ (3 * 16 * WH_LD)   // 12672 bf16 per buffer

#if __has_builtin(__builtin_amdgcn_global_load_async_to_lds_b128) && \
    __has_builtin(__builtin_amdgcn_s_wait_asynccnt)
#define USE_ASYNC 1
#else
#define USE_ASYNC 0
#endif

// 16-byte global -> LDS copy (async on gfx1250 when the builtin exists)
__device__ __forceinline__ void copy16_g2l(const void* src, void* dst) {
#if USE_ASYNC
    __builtin_amdgcn_global_load_async_to_lds_b128(
        (__attribute__((address_space(1))) v4i*)(unsigned long long)src,
        (__attribute__((address_space(3))) v4i*)dst,
        0, 0);
#else
    *(uint4*)dst = *(const uint4*)src;
#endif
}
__device__ __forceinline__ void wait_g2l() {
#if USE_ASYNC
    __builtin_amdgcn_s_wait_asynccnt(0);
#endif
}

// ---- WMMA operand loaders (layouts per CDNA5 ISA 7.12.2) -------------------
// A-matrix 16x32 bf16: lane (hh = l>>4, m = l&15):
//   VGPR0-3: K = k0 + 8*hh + {0..7}      -> bytes 2*k0 + 16*hh
//   VGPR4-7: K = k0 + 16 + 8*hh + {0..7} -> bytes 2*k0 + 16*hh + 32
__device__ __forceinline__ v16bf lda(const __bf16* p, int hh) {
    const char* c = (const char*)p + 16 * hh;
    v16bf r;
    ((uint4*)&r)[0] = *(const uint4*)(c);
    ((uint4*)&r)[1] = *(const uint4*)(c + 32);
    return r;
}
// B-matrix 32x16 bf16: lane (hh, n = l&15) holds K = k0 + 8*hh + {0..7},
// k0 + 16 + 8*hh + {0..7} of column n (same K order as A); B = W^T stored as rows of W.
__device__ __forceinline__ v16bf ldb(const __bf16* p, int hh) {
    const char* c = (const char*)p + 16 * hh;
    v16bf r;
    ((uint4*)&r)[0] = *(const uint4*)(c);
    ((uint4*)&r)[1] = *(const uint4*)(c + 32);
    return r;
}

__device__ __forceinline__ v8f wmma_bf16(v16bf a, v16bf b, v8f c) {
    return __builtin_amdgcn_wmma_f32_16x16x32_bf16(
        false, a, false, b, (short)0, c, false, false);
}

__device__ __forceinline__ float sigmoid_f(float x) {
    return 1.0f / (1.0f + __expf(-x));
}
__device__ __forceinline__ float tanh_f(float x) {
    float e = __expf(2.0f * x);
    return (e - 1.0f) / (e + 1.0f);
}

// ---- Stage 0: convert weights to bf16 in workspace -------------------------
__global__ void gru_cvt_weights(const float* __restrict__ wp,
                                const float* __restrict__ wih,
                                const float* __restrict__ whh,
                                __bf16* __restrict__ wpb,
                                __bf16* __restrict__ wihb,
                                __bf16* __restrict__ whhb) {
    int i = blockIdx.x * blockDim.x + threadIdx.x;
    if (i < PP * 64) {                       // W_proj [128][40] -> [128][64] padded
        int r = i >> 6, c = i & 63;
        wpb[i] = (c < INF) ? (__bf16)wp[r * INF + c] : (__bf16)0.0f;
    }
    if (i < 3 * HH * PP) wihb[i] = (__bf16)wih[i];   // [768][128]
    if (i < 3 * HH * HH) whhb[i] = (__bf16)whh[i];   // [768][256]
}

// ---- Fused GRU step --------------------------------------------------------
__global__ __launch_bounds__(NTHR)
void gru_step_kernel(const float* __restrict__ x_step,
                     const float* __restrict__ hidden,
                     const float* __restrict__ b_proj,
                     const float* __restrict__ b_ih,
                     const float* __restrict__ b_hh,
                     const float* __restrict__ W_head,
                     const float* __restrict__ b_head,
                     const __bf16* __restrict__ wpb,
                     const __bf16* __restrict__ wihb,
                     const __bf16* __restrict__ whhb,
                     float* __restrict__ pred,
                     float* __restrict__ hout) {
    __shared__ __bf16 sx [BT * XA_LD];    // x tile, bf16, K-padded
    __shared__ __bf16 shd[BT * HA_LD];    // h tile, bf16
    __shared__ __bf16 sxp[BT * XP_LD];    // x_proj tile, bf16
    __shared__ __bf16 swi[2 * WIB_SZ];    // W_ih N-tile double buffer
    __shared__ __bf16 swh[2 * WHB_SZ];    // W_hh N-tile double buffer

    const int tid  = threadIdx.x;
    const int b0   = blockIdx.x * BT;
    const int wv   = tid >> 5;
    const int lane = tid & 31;
    const int hh   = lane >> 4;
    const int ln   = lane & 15;
    const int m0   = wv * 16;

    // ---- stage weight N-tile `nt` into LDS buffer (coalesced 16B chunks) ----
    // W_ih: 3 gates x 16 rows x 256B = 768 chunks; W_hh: 3 x 16 x 512B = 1536.
    auto stage_w = [&](int nt, int buf) {
        __bf16* dwi = swi + buf * WIB_SZ;
        __bf16* dwh = swh + buf * WHB_SZ;
#pragma unroll
        for (int c = tid; c < 768; c += NTHR) {
            int g = c >> 8, rem = c & 255, row = rem >> 4, off = rem & 15;
            const char* s = (const char*)(wihb + (size_t)((g << 8) + nt * 16 + row) * PP) + off * 16;
            char* d = (char*)(dwi + (g * 16 + row) * WI_LD) + off * 16;
            copy16_g2l(s, d);
        }
#pragma unroll
        for (int c = tid; c < 1536; c += NTHR) {
            int g = c >> 9, rem = c & 511, row = rem >> 5, off = rem & 31;
            const char* s = (const char*)(whhb + (size_t)((g << 8) + nt * 16 + row) * HH) + off * 16;
            char* d = (char*)(dwh + (g * 16 + row) * WH_LD) + off * 16;
            copy16_g2l(s, d);
        }
    };

    // Prefetch weight tile nt=0 immediately; overlaps phases 0-1.
    stage_w(0, 0);

    // ---- Phase 0: stage x (zero-padded) and h into LDS as bf16 ----
    const float* xsrc = x_step + (size_t)b0 * INF;
    for (int i = tid; i < BT * 64; i += NTHR) {
        int r = i >> 6, c = i & 63;
        float v = (c < INF) ? xsrc[r * INF + c] : 0.0f;
        sx[r * XA_LD + c] = (__bf16)v;
    }
    const float* hsrc = hidden + (size_t)b0 * HH;
    for (int i = tid; i < BT * (HH / 4); i += NTHR) {
        int r = i >> 6;
        int c = (i & 63) << 2;
        const float4 v = *reinterpret_cast<const float4*>(hsrc + r * HH + c);
        bf4 o;
        o.x = (__bf16)v.x; o.y = (__bf16)v.y; o.z = (__bf16)v.z; o.w = (__bf16)v.w;
        *reinterpret_cast<bf4*>(&shd[r * HA_LD + c]) = o;
    }
    __syncthreads();

    // ---- Phase 1: x_proj = x @ W_proj^T + b_proj (WMMA, K padded to 64) ----
    const __bf16* xarow = &sx[(m0 + ln) * XA_LD];
    for (int nt = 0; nt < PP / 16; ++nt) {
        v8f acc = {0.f, 0.f, 0.f, 0.f, 0.f, 0.f, 0.f, 0.f};
        const __bf16* bbase = wpb + (nt * 16 + ln) * 64;
#pragma unroll
        for (int k0 = 0; k0 < 64; k0 += 32) {
            v16bf a = lda(xarow + k0, hh);
            v16bf b = ldb(bbase + k0, hh);
            acc = wmma_bf16(a, b, acc);
        }
        float bp = b_proj[nt * 16 + ln];
#pragma unroll
        for (int v = 0; v < 8; ++v) {
            int row = m0 + v + 8 * hh;
            sxp[row * XP_LD + nt * 16 + ln] = (__bf16)(acc[v] + bp);
        }
    }

    // ---- Phase 2: gates + elementwise + head partials ----
    float pacc[8];
#pragma unroll
    for (int v = 0; v < 8; ++v) pacc[v] = 0.0f;

    const __bf16* xprow = &sxp[(m0 + ln) * XP_LD];
    const __bf16* harow = &shd[(m0 + ln) * HA_LD];

    for (int nt = 0; nt < HH / 16; ++nt) {
        const int cur = nt & 1;
        wait_g2l();          // own async fills of buffer `cur` complete
        __syncthreads();     // everyone's fills done; prior reads of other buf done
        if (nt + 1 < HH / 16) stage_w(nt + 1, cur ^ 1);

        const __bf16* wi = swi + cur * WIB_SZ;
        const __bf16* wh = swh + cur * WHB_SZ;
        const int col = nt * 16 + ln;

        v8f gir = {0.f,0.f,0.f,0.f,0.f,0.f,0.f,0.f};
        v8f giz = gir, gin = gir, ghr = gir, ghz = gir, ghn = gir;

        // gi = x_proj @ W_ih^T   (K = 128); A shared across the 3 gates
#pragma unroll
        for (int k0 = 0; k0 < PP; k0 += 32) {
            v16bf a = lda(xprow + k0, hh);
            gir = wmma_bf16(a, ldb(wi + (0 * 16 + ln) * WI_LD + k0, hh), gir);
            giz = wmma_bf16(a, ldb(wi + (1 * 16 + ln) * WI_LD + k0, hh), giz);
            gin = wmma_bf16(a, ldb(wi + (2 * 16 + ln) * WI_LD + k0, hh), gin);
        }
        // gh = h @ W_hh^T        (K = 256)
#pragma unroll
        for (int k0 = 0; k0 < HH; k0 += 32) {
            v16bf a = lda(harow + k0, hh);
            ghr = wmma_bf16(a, ldb(wh + (0 * 16 + ln) * WH_LD + k0, hh), ghr);
            ghz = wmma_bf16(a, ldb(wh + (1 * 16 + ln) * WH_LD + k0, hh), ghz);
            ghn = wmma_bf16(a, ldb(wh + (2 * 16 + ln) * WH_LD + k0, hh), ghn);
        }

        const float bihr = b_ih[col], bihz = b_ih[HH + col], bihn = b_ih[2 * HH + col];
        const float bhhr = b_hh[col], bhhz = b_hh[HH + col], bhhn = b_hh[2 * HH + col];
        const float whd  = W_head[col];

#pragma unroll
        for (int v = 0; v < 8; ++v) {
            const int row = b0 + m0 + v + 8 * hh;
            float r = sigmoid_f(gir[v] + bihr + ghr[v] + bhhr);
            float z = sigmoid_f(giz[v] + bihz + ghz[v] + bhhz);
            float n = tanh_f(gin[v] + bihn + r * (ghn[v] + bhhn));
            float hprev = hidden[(size_t)row * HH + col];
            float hn = (1.0f - z) * n + z * hprev;
            hout[(size_t)row * HH + col] = hn;
            pacc[v] += hn * whd;
        }
    }

    // ---- Phase 3: head reduction across the 16 lanes of each half ----
    const float bh = b_head[0];
#pragma unroll
    for (int v = 0; v < 8; ++v) {
        float s = pacc[v];
        for (int off = 8; off >= 1; off >>= 1) s += __shfl_xor(s, off, 16);
        if (ln == 0) pred[b0 + m0 + v + 8 * hh] = s + bh;
    }
}

extern "C" void kernel_launch(void* const* d_in, const int* in_sizes, int n_in,
                              void* d_out, int out_size, void* d_ws, size_t ws_size,
                              hipStream_t stream) {
    const float* x_step = (const float*)d_in[0];   // [B,1,40]
    const float* hidden = (const float*)d_in[1];   // [1,B,256]
    const float* W_proj = (const float*)d_in[2];   // [128,40]
    const float* b_proj = (const float*)d_in[3];   // [128]
    const float* W_ih   = (const float*)d_in[4];   // [768,128]
    const float* W_hh   = (const float*)d_in[5];   // [768,256]
    const float* b_ih   = (const float*)d_in[6];   // [768]
    const float* b_hh   = (const float*)d_in[7];   // [768]
    const float* W_head = (const float*)d_in[8];   // [1,256]
    const float* b_head = (const float*)d_in[9];   // [1]

    float* pred = (float*)d_out;          // [B,1,1] flat = B floats
    float* hout = pred + BATCH;           // [1,B,256] flat

    __bf16* wpb  = (__bf16*)d_ws;                   // 128*64
    __bf16* wihb = wpb + PP * 64;                   // 768*128
    __bf16* whhb = wihb + 3 * HH * PP;              // 768*256
    (void)in_sizes; (void)n_in; (void)out_size; (void)ws_size;

    gru_cvt_weights<<<(3 * HH * HH + 255) / 256, 256, 0, stream>>>(
        W_proj, W_ih, W_hh, wpb, wihb, whhb);

    gru_step_kernel<<<BATCH / BT, NTHR, 0, stream>>>(
        x_step, hidden, b_proj, b_ih, b_hh, W_head, b_head,
        wpb, wihb, whhb, pred, hout);
}
